// ContrastiveLoss_47038481826366
// MI455X (gfx1250) — compile-verified, matrix operand order checked by reading, not run on hardware
//
#include <hip/hip_runtime.h>
#include <hip/hip_bf16.h>

typedef __attribute__((ext_vector_type(16))) _Float16 v16h;
typedef __attribute__((ext_vector_type(8)))  _Float16 h8;
typedef __attribute__((ext_vector_type(4)))  _Float16 h4;
typedef __attribute__((ext_vector_type(8)))  float    v8f;

#define NROW 8192
#define DDIM 1024
#define INV_T 10.0f
#define LMAX 10.0f                 // cos<=1 -> logits <= 10: fixed LSE shift
#define MASKED_LOGIT (-65504.0f * 10.0f)

#define SHUF16(lo, hi) __builtin_shufflevector(lo, hi, 0,1,2,3,4,5,6,7,8,9,10,11,12,13,14,15)

// ---------------- Kernel A: row-normalize z (f32) -> zhat (f16) ----------------
__global__ __launch_bounds__(256) void normalize_f16(const float* __restrict__ z,
                                                     _Float16* __restrict__ zh) {
  const int r = blockIdx.x;
  const int t = threadIdx.x;
  const float4 v = *(const float4*)(z + (size_t)r * DDIM + t * 4);
  float s = v.x * v.x + v.y * v.y + v.z * v.z + v.w * v.w;
  __shared__ float red[256];
  red[t] = s;
  __syncthreads();
  for (int o = 128; o > 0; o >>= 1) {
    if (t < o) red[t] += red[t + o];
    __syncthreads();
  }
  const float inv = rsqrtf(fmaxf(red[0], 1e-24f));
  h4 o;
  o[0] = (_Float16)(v.x * inv);
  o[1] = (_Float16)(v.y * inv);
  o[2] = (_Float16)(v.z * inv);
  o[3] = (_Float16)(v.w * inv);
  *(h4*)(zh + (size_t)r * DDIM + t * 4) = o;
}

// ------ Kernel B: fused  zhat @ zhat^T  +  mask/temp  +  fixed-shift LSE + pos ------
// 128 blocks x 128 threads (4 waves). Wave w owns rows [block*64 + w*16, +16), with
// its full 16x1024 A strip cached in 256 VGPRs (v16h afrag[32]) for the whole sweep.
// Two column tiles are processed per iteration: two independent WMMA chains and two
// independent B load streams so loads and matrix ops from the other stream cover
// each other's latency.
__global__ __launch_bounds__(128) void simclr_gemm_lse(const _Float16* __restrict__ zh,
                                                       float* __restrict__ nll) {
  const int t    = threadIdx.x;
  const int wave = t >> 5;
  const int lane = t & 31;
  const int n16  = lane & 15;                   // A row index / B col index within tile
  const int hsel = (lane >> 4) << 3;            // 0 or 8: K-chunk half; also C row offset
  const int rbase2 = blockIdx.x * 64 + wave * 16 + hsel;   // this slot-octet's row base
  const int pbase  = (rbase2 + 4096) & (NROW - 1);         // positive-col base (no wrap in +v)

  // --- A register cache: one load of this wave's strip, WMMA fragment layout ---
  const _Float16* Arow = zh + (size_t)(blockIdx.x * 64 + wave * 16 + n16) * DDIM + hsel;
  v16h afrag[32];
#pragma unroll
  for (int ks = 0; ks < 32; ks++) {
    h8 lo = *(const h8*)(Arow + ks * 32);
    h8 hi = *(const h8*)(Arow + ks * 32 + 16);
    afrag[ks] = SHUF16(lo, hi);
  }

  float se[8], pa[8];                           // sum of exp(l-LMAX); positive logit
#pragma unroll
  for (int v = 0; v < 8; v++) { se[v] = 0.0f; pa[v] = 0.0f; }

  for (int ct = 0; ct < 256; ++ct) {
    const int c0 = ct * 32 + n16;               // stream-0 output column
    const int c1 = c0 + 16;                     // stream-1 output column
    const _Float16* B0 = zh + (size_t)c0 * DDIM + hsel;
    const _Float16* B1 = zh + (size_t)c1 * DDIM + hsel;

    // Per-stream 2-deep prefetch rotation; full unroll makes the guard static.
    h8 p0l = *(const h8*)(B0 +  0), p0h = *(const h8*)(B0 + 16);
    h8 q0l = *(const h8*)(B1 +  0), q0h = *(const h8*)(B1 + 16);
    h8 p1l = *(const h8*)(B0 + 32), p1h = *(const h8*)(B0 + 48);
    h8 q1l = *(const h8*)(B1 + 32), q1h = *(const h8*)(B1 + 48);
    v8f acc0 = {}, acc1 = {};
#pragma unroll
    for (int ks = 0; ks < 32; ks++) {
      h8 npl = p1l, nph = p1h, nql = q1l, nqh = q1h;
      if (ks < 30) {
        npl = *(const h8*)(B0 + (ks + 2) * 32);
        nph = *(const h8*)(B0 + (ks + 2) * 32 + 16);
        nql = *(const h8*)(B1 + (ks + 2) * 32);
        nqh = *(const h8*)(B1 + (ks + 2) * 32 + 16);
      }
      acc0 = __builtin_amdgcn_wmma_f32_16x16x32_f16(false, afrag[ks],
                                                    false, SHUF16(p0l, p0h),
                                                    (short)0, acc0, false, false);
      acc1 = __builtin_amdgcn_wmma_f32_16x16x32_f16(false, afrag[ks],
                                                    false, SHUF16(q0l, q0h),
                                                    (short)0, acc1, false, false);
      p0l = p1l; p0h = p1h; p1l = npl; p1h = nph;
      q0l = q1l; q0h = q1h; q1l = nql; q1h = nqh;
    }

    // Branchless epilogue. C/D layout: VGPR v, lanes 0-15: M=v ; lanes 16-31: M=v+8.
    const int dc0  = c0 - rbase2;               // == v on the diagonal (stream 0)
    const int dcp0 = c0 - pbase;                // == v on the positive column (stream 0)
#pragma unroll
    for (int v = 0; v < 8; v++) {
      const float l0 = (dc0 == v)      ? MASKED_LOGIT : acc0[v] * INV_T;
      const float l1 = (dc0 - 16 == v) ? MASKED_LOGIT : acc1[v] * INV_T;
      se[v] += __expf(l0 - LMAX) + __expf(l1 - LMAX);
      pa[v] += (dcp0 == v)      ? l0 : 0.0f;    // exactly one hit per row overall
      pa[v] += (dcp0 - 16 == v) ? l1 : 0.0f;
    }
  }

  // Sum across the 16 lanes that hold each row (xor offsets stay within a half).
#pragma unroll
  for (int off = 1; off < 16; off <<= 1) {
#pragma unroll
    for (int v = 0; v < 8; v++) {
      se[v] += __shfl_xor(se[v], off, 32);
      pa[v] += __shfl_xor(pa[v], off, 32);
    }
  }
  if (n16 == 0) {
#pragma unroll
    for (int v = 0; v < 8; v++) {
      nll[rbase2 + v] = -pa[v] + LMAX + __logf(se[v]);
    }
  }
}

// ---------------- Kernel C: mean over the 8192 per-row NLLs ----------------
__global__ __launch_bounds__(256) void mean_reduce(const float* __restrict__ nll,
                                                   float* __restrict__ out) {
  __shared__ float red[256];
  float s = 0.0f;
  for (int i = threadIdx.x; i < NROW; i += 256) s += nll[i];
  red[threadIdx.x] = s;
  __syncthreads();
  for (int o = 128; o > 0; o >>= 1) {
    if (threadIdx.x < o) red[threadIdx.x] += red[threadIdx.x + o];
    __syncthreads();
  }
  if (threadIdx.x == 0) out[0] = red[0] / (float)NROW;
}

extern "C" void kernel_launch(void* const* d_in, const int* in_sizes, int n_in,
                              void* d_out, int out_size, void* d_ws, size_t ws_size,
                              hipStream_t stream) {
  (void)in_sizes; (void)n_in; (void)out_size; (void)ws_size;
  const float* z = (const float*)d_in[0];
  float* out = (float*)d_out;
  _Float16* zh = (_Float16*)d_ws;                                     // 16 MB
  float* nll = (float*)((char*)d_ws + (size_t)NROW * DDIM * sizeof(_Float16));

  normalize_f16<<<NROW, 256, 0, stream>>>(z, zh);
  simclr_gemm_lse<<<NROW / 64, 128, 0, stream>>>(zh, nll);
  mean_reduce<<<1, 256, 0, stream>>>(nll, out);
}
